// EdgeGATConv_79920751444086
// MI455X (gfx1250) — matrix operand order repaired and checked by reading, not compiled, not run on hardware
//
#include <hip/hip_runtime.h>
#include <hip/hip_bf16.h>

typedef __attribute__((ext_vector_type(16))) _Float16 v16h;
typedef __attribute__((ext_vector_type(8)))  float    v8f;

#define H_HEADS 4
#define IN_DIM  128
#define OUT_DIM 64
#define EDGE_DIM 32
#define NB      32
#define BATCH   256
#define BM_PER_WG 8

#define ADJ32_S 129   // f32 row stride (pad) for 32x128 tile (hsrc kernel)
#define ADJ16_S 136   // f16 row stride (pad) for 32x128 tile
#define HDST_S  65    // f32 row stride for 32x64 h_dst tile

#define NEG_INF_F (-9.0e15f)
#define LEAKY 0.2f

// ---- WMMA fragment helpers (wave32, 16x16x32 f16) ----------------------

// A tile (16x32 f16), row-major in LDS with stride `stride` halves.
__device__ __forceinline__ v16h load_a_frag(const _Float16* base, int stride,
                                            int row0, int k0, int lane) {
  int hs  = lane >> 4;
  int row = row0 + (lane & 15);
  const _Float16* p0 = base + row * stride + k0 + hs * 8;
  const _Float16* p1 = p0 + 16;
  v16h r;
#pragma unroll
  for (int j = 0; j < 8; ++j) { r[j] = p0[j]; r[8 + j] = p1[j]; }
  return r;
}

// B tile (32x16 f16) from transposed weights wt[o][i] (stride sk halves).
__device__ __forceinline__ v16h load_b_frag(const _Float16* wt, int sk,
                                            int n0, int k0, int lane) {
  int hs = lane >> 4;
  int n  = n0 + (lane & 15);
  const _Float16* p = wt + n * sk + k0 + hs * 8;
  v16h r;
#pragma unroll
  for (int j = 0; j < 8; ++j) { r[j] = p[j]; r[8 + j] = p[16 + j]; }
  return r;
}

// C/D tile (16x16 f32): lane holds col c0+(lane&15); VGPR r -> row r0+r+8*(lane>=16).
__device__ __forceinline__ void store_c_frag(float* dst, int stride,
                                             int r0, int c0, v8f c, int lane) {
  int hs  = lane >> 4;
  int col = c0 + (lane & 15);
#pragma unroll
  for (int r = 0; r < 8; ++r)
    dst[(r0 + r + hs * 8) * stride + col] = c[r];
}

// ---- Kernel 0: weight transpose->f16 + folded attention vectors --------
__global__ void gat_setup_kernel(const float* __restrict__ W_src,
                                 const float* __restrict__ W_dst,
                                 const float* __restrict__ W_edge,
                                 const float* __restrict__ attns,
                                 _Float16* __restrict__ wsrcT,
                                 _Float16* __restrict__ wdstT,
                                 float* __restrict__ wa_src,
                                 float* __restrict__ wa_dst,
                                 float* __restrict__ wa_ed) {
  int tid   = blockIdx.x * blockDim.x + threadIdx.x;
  int tstep = gridDim.x * blockDim.x;

  for (int idx = tid; idx < H_HEADS * OUT_DIM * IN_DIM; idx += tstep) {
    int i = idx % IN_DIM;
    int o = (idx / IN_DIM) % OUT_DIM;
    int h = idx / (IN_DIM * OUT_DIM);
    wsrcT[idx] = (_Float16)W_src[(h * IN_DIM + i) * OUT_DIM + o];
    wdstT[idx] = (_Float16)W_dst[(h * IN_DIM + i) * OUT_DIM + o];
  }
  for (int idx = tid; idx < H_HEADS * IN_DIM; idx += tstep) {
    int i = idx % IN_DIM, h = idx / IN_DIM;
    float ss = 0.f, sd = 0.f;
    for (int o = 0; o < OUT_DIM; ++o) {
      ss += W_src[(h * IN_DIM + i) * OUT_DIM + o] * attns[h * 3 * OUT_DIM + o];
      sd += W_dst[(h * IN_DIM + i) * OUT_DIM + o] * attns[h * 3 * OUT_DIM + OUT_DIM + o];
    }
    wa_src[idx] = ss;
    wa_dst[idx] = sd;
  }
  for (int idx = tid; idx < H_HEADS * EDGE_DIM; idx += tstep) {
    int e = idx % EDGE_DIM, h = idx / EDGE_DIM;
    float se = 0.f;
    for (int o = 0; o < OUT_DIM; ++o)
      se += W_edge[(h * EDGE_DIM + e) * OUT_DIM + o] * attns[h * 3 * OUT_DIM + 2 * OUT_DIM + o];
    wa_ed[idx] = se;
  }
}

// ---- Kernel 1: h_src (WMMA) + s_src, one WG per batch b ----------------
__global__ void __launch_bounds__(128)
gat_hsrc_kernel(const float* __restrict__ x,
                const _Float16* __restrict__ wsrcT,
                const float* __restrict__ wa_src,
                float* __restrict__ hsrc,      // [H][B][N][64] f32
                float* __restrict__ ssrc) {    // [H][B*N] f32
  __shared__ float    xs[NB * ADJ32_S];
  __shared__ _Float16 xh[NB * ADJ16_S];

  int b    = blockIdx.x;
  int tid  = threadIdx.x;
  int wave = tid >> 5, lane = tid & 31;

  const float* xb = x + (size_t)b * NB * IN_DIM;
#pragma unroll
  for (int j = 0; j < 8; ++j) {                 // 1024 float4 / 128 threads
    int f = tid + j * 128;
    float4 v = ((const float4*)xb)[f];
    int n = f >> 5, c = (f & 31) << 2;
    float* d = &xs[n * ADJ32_S + c];
    d[0] = v.x; d[1] = v.y; d[2] = v.z; d[3] = v.w;
    _Float16* dh = &xh[n * ADJ16_S + c];
    dh[0] = (_Float16)v.x; dh[1] = (_Float16)v.y;
    dh[2] = (_Float16)v.z; dh[3] = (_Float16)v.w;
  }
  __syncthreads();

  int h = wave;
  const _Float16* wt = wsrcT + h * OUT_DIM * IN_DIM;
  float* dst = hsrc + (size_t)(h * BATCH + b) * NB * OUT_DIM;

  for (int mt = 0; mt < 2; ++mt)
    for (int nt = 0; nt < 4; ++nt) {
      v8f c = {};
#pragma unroll
      for (int kk = 0; kk < 4; ++kk) {
        v16h a  = load_a_frag(xh, ADJ16_S, mt * 16, kk * 32, lane);
        v16h bb = load_b_frag(wt, IN_DIM, nt * 16, kk * 32, lane);
        c = __builtin_amdgcn_wmma_f32_16x16x32_f16(false, a, false, bb,
                                                   (short)0, c, false, false);
      }
      int hs = lane >> 4;
      int col = nt * 16 + (lane & 15);
#pragma unroll
      for (int r = 0; r < 8; ++r) {
        int row = mt * 16 + r + hs * 8;
        dst[(size_t)row * OUT_DIM + col] = c[r];
      }
    }

  { // s_src[h, b*32+n] = x[b,n,:] . (W_src[h] @ a_src[h])
    int n = lane;
    const float* wa = wa_src + h * IN_DIM;
    float acc = 0.f;
#pragma unroll 8
    for (int i = 0; i < IN_DIM; ++i) acc += xs[n * ADJ32_S + i] * wa[i];
    ssrc[(h * BATCH + b) * NB + n] = acc;
  }
}

// ---- Kernel 2: main — 8 (b,m) tiles per WG, double-buffered ------------
__global__ void __launch_bounds__(128)
gat_main_kernel(const float* __restrict__ adj_x,
                const int*   __restrict__ adj_masks,
                const float* __restrict__ edge_attr,
                const _Float16* __restrict__ wdstT,
                const float* __restrict__ wa_dst,
                const float* __restrict__ wa_ed,
                const float* __restrict__ hsrc,
                const float* __restrict__ ssrc,
                float* __restrict__ out) {
  __shared__ _Float16 axh[2][NB * ADJ16_S];   // ping-pong f16 adj tile
  __shared__ float    hd[H_HEADS][NB * HDST_S];
  __shared__ float    wsm[H_HEADS][NB];

  int tid  = threadIdx.x;
  int wave = tid >> 5, lane = tid & 31;
  int bm0  = blockIdx.x * BM_PER_WG;

  int h = wave;                          // one head per wave
  const _Float16* wt = wdstT + h * OUT_DIM * IN_DIM;  // B-frags loop-invariant

  float4 st[8];                          // staging registers (32 floats/thread)

  auto issue = [&](int it) {             // global loads in flight
    const float* axg = adj_x + (size_t)(bm0 + it) * NB * IN_DIM;
#pragma unroll
    for (int j = 0; j < 8; ++j) st[j] = ((const float4*)axg)[tid + j * 128];
  };
  auto commit = [&](int it) {            // cvt f16 + LDS store
    _Float16* buf = axh[it & 1];
#pragma unroll
    for (int j = 0; j < 8; ++j) {
      int f = tid + j * 128;
      int n = f >> 5, c = (f & 31) << 2;
      _Float16* dh = &buf[n * ADJ16_S + c];
      dh[0] = (_Float16)st[j].x; dh[1] = (_Float16)st[j].y;
      dh[2] = (_Float16)st[j].z; dh[3] = (_Float16)st[j].w;
    }
  };

  issue(0);
  commit(0);
  __syncthreads();

  for (int it = 0; it < BM_PER_WG; ++it) {
    int bm = bm0 + it;
    int b = bm >> 5, m = bm & 31;
    const _Float16* buf = axh[it & 1];

    if (it + 1 < BM_PER_WG) issue(it + 1);   // overlap next tile's HBM latency

    // h_dst[h] = adj_tile (32x128) @ W_dst[h] (128x64), f16 WMMA, f32 acc
    for (int mt = 0; mt < 2; ++mt)
      for (int nt = 0; nt < 4; ++nt) {
        v8f c = {};
#pragma unroll
        for (int kk = 0; kk < 4; ++kk) {
          v16h a  = load_a_frag(buf, ADJ16_S, mt * 16, kk * 32, lane);
          v16h bb = load_b_frag(wt, IN_DIM, nt * 16, kk * 32, lane);
          c = __builtin_amdgcn_wmma_f32_16x16x32_f16(false, a, false, bb,
                                                     (short)0, c, false, false);
        }
        store_c_frag(hd[h], HDST_S, mt * 16, nt * 16, c, lane);
      }

    // scores + leaky + mask + wave32 softmax: lane == n
    {
      int n = lane;
      const _Float16* ar = &buf[n * ADJ16_S];
      const float* wa = wa_dst + h * IN_DIM;
      float sd = 0.f;
#pragma unroll 8
      for (int i = 0; i < IN_DIM; ++i) sd += (float)ar[i] * wa[i];

      const float* er = edge_attr + ((size_t)bm * NB + n) * EDGE_DIM;
      const float* we = wa_ed + h * EDGE_DIM;
      float se = 0.f;
#pragma unroll
      for (int e2 = 0; e2 < EDGE_DIM; ++e2) se += er[e2] * we[e2];

      // reference tile quirk: s_src_t[h,b,m,n] = s_src[h, (b%8)*32+m, n]
      float stc = ssrc[(h * BATCH + ((b & 7) * 32 + m)) * NB + n];
      float s = stc + sd + se;
      s = (s > 0.f) ? s : LEAKY * s;
      if (adj_masks[bm * NB + n] == 0) s = NEG_INF_F;

      float mx = s;
#pragma unroll
      for (int off = 16; off > 0; off >>= 1) mx = fmaxf(mx, __shfl_xor(mx, off, 32));
      float e = __expf(s - mx);
      float sum = e;
#pragma unroll
      for (int off = 16; off > 0; off >>= 1) sum += __shfl_xor(sum, off, 32);
      wsm[h][n] = e / sum;
    }

    if (it + 1 < BM_PER_WG) commit(it + 1);  // publish next tile
    __syncthreads();                         // next buf staged; hd/wsm settled

    // out[b,m,h*64+o] = h_src[h,b,m,o] + sum_n w[h,n] * h_dst[h,n,o]
    {
      const float* hsr = hsrc + ((size_t)(h * BATCH + b) * NB + m) * OUT_DIM;
#pragma unroll
      for (int t = 0; t < 2; ++t) {
        int o = lane + t * 32;
        float acc = hsr[o];
#pragma unroll
        for (int n = 0; n < NB; ++n) acc += wsm[h][n] * hd[h][n * HDST_S + o];
        out[(size_t)bm * (H_HEADS * OUT_DIM) + h * OUT_DIM + o] = acc;
      }
    }
  }
}

// ---- launch ------------------------------------------------------------
extern "C" void kernel_launch(void* const* d_in, const int* in_sizes, int n_in,
                              void* d_out, int out_size, void* d_ws, size_t ws_size,
                              hipStream_t stream) {
  (void)in_sizes; (void)n_in; (void)out_size; (void)ws_size;
  const float* x        = (const float*)d_in[0];
  const float* adj_x    = (const float*)d_in[1];
  const int*   adj_m    = (const int*)d_in[2];
  const float* edge     = (const float*)d_in[3];
  const float* W_src    = (const float*)d_in[4];
  const float* W_dst    = (const float*)d_in[5];
  const float* W_edge   = (const float*)d_in[6];
  const float* attns    = (const float*)d_in[7];
  float* out = (float*)d_out;

  // workspace carve-up
  _Float16* wsrcT = (_Float16*)d_ws;                         // 4*64*128 f16
  _Float16* wdstT = wsrcT + H_HEADS * OUT_DIM * IN_DIM;      // 4*64*128 f16
  float* wa_src = (float*)(wdstT + H_HEADS * OUT_DIM * IN_DIM);
  float* wa_dst = wa_src + H_HEADS * IN_DIM;
  float* wa_ed  = wa_dst + H_HEADS * IN_DIM;
  float* hsrc   = wa_ed + H_HEADS * EDGE_DIM;                // 4*256*32*64 f32 (8 MB)
  float* ssrc   = hsrc + (size_t)H_HEADS * BATCH * NB * OUT_DIM; // 4*8192 f32

  gat_setup_kernel<<<64, 256, 0, stream>>>(W_src, W_dst, W_edge, attns,
                                           wsrcT, wdstT, wa_src, wa_dst, wa_ed);
  gat_hsrc_kernel<<<BATCH, 128, 0, stream>>>(x, wsrcT, wa_src, hsrc, ssrc);
  gat_main_kernel<<<(BATCH * NB) / BM_PER_WG, 128, 0, stream>>>(adj_x, adj_m, edge,
                                                                wdstT, wa_dst, wa_ed,
                                                                hsrc, ssrc, out);
}
